// TextSeg_31868657336567
// MI455X (gfx1250) — matrix operand order repaired and checked by reading, not compiled, not run on hardware
//
#include <hip/hip_runtime.h>
#include <cstdint>
#include <cstddef>

// ---------------------------------------------------------------------------
// Model constants
// ---------------------------------------------------------------------------
#define BB   16
#define SS   96
#define WW   48
#define EE   300
#define EPAD 320
#define HH   256
#define GG   1024   // 4*H gates

typedef __attribute__((ext_vector_type(16))) _Float16 v16h;
typedef __attribute__((ext_vector_type(8)))  float    v8f;

union AFrag { v16h v; _Float16 e[16]; uint4 q[2]; };

// Fast transcendentals: gfx1250 has V_TANH_F32
#if __has_builtin(__builtin_amdgcn_tanhf)
__device__ __forceinline__ float tanh_fast(float x) { return __builtin_amdgcn_tanhf(x); }
#elif __has_builtin(__builtin_amdgcn_tanh_f32)
__device__ __forceinline__ float tanh_fast(float x) { return __builtin_amdgcn_tanh_f32(x); }
#else
__device__ __forceinline__ float tanh_fast(float x) { return tanhf(x); }
#endif
__device__ __forceinline__ float sigf(float x) { return 0.5f * tanh_fast(0.5f * x) + 0.5f; }

// ---------------------------------------------------------------------------
// f32 -> f16 conversion of x with K padding (E=300 -> 320)
// ---------------------------------------------------------------------------
__global__ __launch_bounds__(256) void cvt_x_kernel(const float* __restrict__ x,
                                                    _Float16* __restrict__ xh)
{
    size_t i = (size_t)blockIdx.x * blockDim.x + threadIdx.x;
    size_t total = (size_t)BB * SS * WW * EPAD;
    if (i >= total) return;
    int    k   = (int)(i % EPAD);
    size_t row = i / EPAD;
    xh[i] = (k < EE) ? (_Float16)x[row * EE + k] : (_Float16)0.0f;
}

// ---------------------------------------------------------------------------
// Weight conversion: f32 (1024 x Kin) -> f16 (1024 x Kp), row-permuted so that
// new row n' = 4*j + g  comes from  old row g*256 + j   (gate interleave)
// ---------------------------------------------------------------------------
__global__ __launch_bounds__(256) void cvt_w_kernel(const float* __restrict__ w,
                                                    _Float16* __restrict__ dst,
                                                    int Kin, int Kp)
{
    int np = blockIdx.x;                       // 0..1023 (new row)
    int srcRow = (np & 3) * 256 + (np >> 2);
    for (int k = threadIdx.x; k < Kp; k += blockDim.x)
        dst[(size_t)np * Kp + k] = (k < Kin) ? (_Float16)w[(size_t)srcRow * Kin + k]
                                             : (_Float16)0.0f;
}

__global__ __launch_bounds__(256) void cvt_b_kernel(const float* __restrict__ b,
                                                    float* __restrict__ dst)
{
    int np = blockIdx.x * blockDim.x + threadIdx.x;
    if (np < GG) dst[np] = b[(np & 3) * 256 + (np >> 2)];
}

// ---------------------------------------------------------------------------
// pooled (B,S,512) f32 -> masked f16 (zero where s >= doc_len[b])
// ---------------------------------------------------------------------------
__global__ __launch_bounds__(256) void cvt_pool_kernel(const float* __restrict__ pooled,
                                                       const int* __restrict__ dl,
                                                       _Float16* __restrict__ ph)
{
    int i = blockIdx.x * blockDim.x + threadIdx.x;
    if (i >= BB * SS * 512) return;
    int t = i >> 9;          // token index b*S + s
    int s = t % SS;
    int b = t / SS;
    ph[i] = (s < dl[b]) ? (_Float16)pooled[i] : (_Float16)0.0f;
}

// ---------------------------------------------------------------------------
// One K-tile (K=32) of the 16x128 output slab owned by a wave.
// Breadth-first: all 8 B fragments are loaded into distinct registers (one
// 16-wide b128 load clause), then the 8 WMMAs run while later loads are still
// in flight -> latency amortized across the tile instead of per-WMMA waits.
// All tile offsets are immediates (row stride BSTRIDE is a template const).
// ---------------------------------------------------------------------------
template<int BSTRIDE>
__device__ __forceinline__ void gemm_ktile(const AFrag& a, AFrag& aNext,
                                           const _Float16* __restrict__ apNext,
                                           const _Float16* __restrict__ bp,
                                           v8f* __restrict__ acc, bool loadNext)
{
    AFrag b[8];
#pragma unroll
    for (int t8 = 0; t8 < 8; ++t8) {
        const _Float16* p = bp + (size_t)t8 * 16 * BSTRIDE;
        b[t8].q[0] = *(const uint4*)(p);
        b[t8].q[1] = *(const uint4*)(p + 16);
    }
    if (loadNext) {
        aNext.q[0] = *(const uint4*)(apNext);
        aNext.q[1] = *(const uint4*)(apNext + 16);
    }
#pragma unroll
    for (int t8 = 0; t8 < 8; ++t8) {
        acc[t8] = __builtin_amdgcn_wmma_f32_16x16x32_f16(
            false, a.v, false, b[t8].v, (short)0, acc[t8], false, false);
    }
}

// ---------------------------------------------------------------------------
// Generic single-direction WMMA LSTM (templated on input width KX).
//   block = 256 threads (8 waves), 16 sequences per block (M tile = 16)
//   grid.x = sequence groups, grid.y = direction (0 fwd, 1 bwd)
//   mode 0: max-pool h over valid t -> out[(seqBase+m)*poolStride + dir*256 + j]
//   mode 1: write h series (masked by lens if non-null)
// ---------------------------------------------------------------------------
template<int KX>
__global__ __launch_bounds__(256) void lstm_kernel(
    const _Float16* __restrict__ xin, int T, int xRowStride,
    const _Float16* __restrict__ wB0, const _Float16* __restrict__ uB0, const float* __restrict__ b0,
    const _Float16* __restrict__ wB1, const _Float16* __restrict__ uB1, const float* __restrict__ b1,
    const int* __restrict__ lens, int mode,
    float* __restrict__ out, int outRowStride, int outTStride, int poolStride)
{
    __shared__ alignas(16) _Float16 hsh[16 * 256];   // h state (f16, A-matrix source)
    __shared__ alignas(16) float    csh[16 * 256];   // c state
    __shared__ alignas(16) float    hmax[16 * 256];  // running max-pool (mode 0)
    __shared__ int lsh[16];

    const int tid  = threadIdx.x;
    const int lane = tid & 31;
    const int wave = tid >> 5;
    const int l15  = lane & 15;
    const int lhi  = lane >> 4;
    const int dir  = blockIdx.y;
    const int seqBase = blockIdx.x * 16;

    const _Float16* wB = dir ? wB1 : wB0;
    const _Float16* uB = dir ? uB1 : uB0;
    const float*    bb = dir ? b1  : b0;

    for (int i = tid; i < 16 * 256; i += 256) {
        hsh[i] = (_Float16)0.0f;
        csh[i] = 0.0f;
        hmax[i] = -3.0e38f;
    }
    if (tid < 16) lsh[tid] = lens ? lens[seqBase + tid] : 0x7fffffff;
    __syncthreads();

    // bias per N-tile (same for all 8 C rows of a tile)
    float bv[8];
#pragma unroll
    for (int t8 = 0; t8 < 8; ++t8) bv[t8] = bb[wave * 128 + t8 * 16 + l15];

    constexpr int NKX = KX / 32;
    const _Float16* wBbase = wB + (size_t)(wave * 128 + l15) * KX + (lhi << 3);
    const _Float16* uBbase = uB + (size_t)(wave * 128 + l15) * 256 + (lhi << 3);
    const _Float16* xseq   = xin + (size_t)(seqBase + l15) * xRowStride;

    for (int step = 0; step < T; ++step) {
        const int t = dir ? (T - 1 - step) : step;

        // Opaque zero offset (uniform -> SGPR). Volatile asm executes every
        // iteration, so the weight-fragment addresses are not loop-invariant
        // (no LICM -> no scratch spilling), but the base pointer's address
        // space is preserved (global_load, not flat_load).
        int woff = 0, uoff = 0;
        asm volatile("" : "+s"(woff));
        asm volatile("" : "+s"(uoff));
        const _Float16* wBp = wBbase + woff;
        const _Float16* uBp = uBbase + uoff;

        // prefetch next step's activation row into cache (speculative)
        if (step + 1 < T) {
            const int tn = dir ? (t - 1) : (t + 1);
            const _Float16* pf = xseq + (size_t)tn * KX + lane * 32;
            __builtin_prefetch(pf, 0, 1);
        }

        v8f acc[8];
#pragma unroll
        for (int t8 = 0; t8 < 8; ++t8) {
            v8f z;
#pragma unroll
            for (int r = 0; r < 8; ++r) z[r] = bv[t8];
            acc[t8] = z;
        }

        // ---- input contribution: x_t (16 x KX) @ W^T ----
        {
            const _Float16* apx = xseq + (size_t)t * KX + (lhi << 3);
            const _Float16* bpw = wBp;
            AFrag aC, aN;
            aC.q[0] = *(const uint4*)(apx);
            aC.q[1] = *(const uint4*)(apx + 16);
#pragma unroll 1
            for (int kt = 0; kt < NKX; kt += 2) {
                gemm_ktile<KX>(aC, aN, apx + 32, bpw,      acc, true);
                gemm_ktile<KX>(aN, aC, apx + 64, bpw + 32, acc, kt + 2 < NKX);
                apx += 64; bpw += 64;
            }
        }

        // ---- recurrent contribution: h (16 x 256) @ U^T ----
        {
            const _Float16* aph = &hsh[l15 * 256 + (lhi << 3)];
            const _Float16* bpu = uBp;
            AFrag aC, aN;
            aC.q[0] = *(const uint4*)(aph);
            aC.q[1] = *(const uint4*)(aph + 16);
#pragma unroll 1
            for (int kt = 0; kt < 8; kt += 2) {
                gemm_ktile<256>(aC, aN, aph + 32, bpu,      acc, true);
                gemm_ktile<256>(aN, aC, aph + 64, bpu + 32, acc, kt + 2 < 8);
                aph += 64; bpu += 64;
            }
        }
        __syncthreads();   // all waves done reading previous h

        // ---- elementwise LSTM cell; gates i,f,g,o live in 4 adjacent lanes ----
        const int lbase = lane & ~3;
        const int low2  = lane & 3;
#pragma unroll
        for (int t8 = 0; t8 < 8; ++t8) {
            const int jj = (wave * 128 + t8 * 16 + l15) >> 2;   // hidden index
#pragma unroll
            for (int r = 0; r < 8; ++r) {
                float v  = acc[t8][r];
                float gi = __shfl(v, lbase + 0, 32);
                float gf = __shfl(v, lbase + 1, 32);
                float gg = __shfl(v, lbase + 2, 32);
                float go = __shfl(v, lbase + 3, 32);
                const int m = r + (lhi << 3);
                float c = csh[m * 256 + jj];
                c = sigf(gf) * c + sigf(gi) * tanh_fast(gg);
                float h = sigf(go) * tanh_fast(c);
                if (low2 == 0) {
                    csh[m * 256 + jj] = c;
                    hsh[m * 256 + jj] = (_Float16)h;
                    if (mode == 0) {
                        float hv = (t < lsh[m]) ? h : -3.0e38f;
                        if (hv > hmax[m * 256 + jj]) hmax[m * 256 + jj] = hv;
                    } else {
                        float hv = (t < lsh[m]) ? h : 0.0f;
                        out[(size_t)m * outRowStride + (size_t)t * outTStride
                            + dir * 256 + jj] = hv;
                    }
                }
            }
        }
        __syncthreads();   // h/c updated before next step's reads
    }

    if (mode == 0) {
        for (int i = tid; i < 16 * 256; i += 256) {
            const int m = i >> 8, j = i & 255;
            out[(size_t)(seqBase + m) * poolStride + dir * 256 + j] = hmax[i];
        }
    }
}

// ---------------------------------------------------------------------------
// Restricted attention (WIN = +-3) over hs (B,S,512); emits hr f16 (B,S,1024)
// One block of 256 threads per token (b,s).
// ---------------------------------------------------------------------------
__global__ __launch_bounds__(256) void attn_kernel(const float* __restrict__ hs,
                                                   const int* __restrict__ dl,
                                                   const float* __restrict__ aw,
                                                   const float* __restrict__ ab,
                                                   _Float16* __restrict__ hr)
{
    const int token = blockIdx.x;          // 0 .. B*S-1
    const int b = token / SS;
    const int s = token % SS;
    const int tid = threadIdx.x;

    __shared__ float Xs[512];
    __shared__ float Xw[7][512];
    __shared__ float sums[23];
    __shared__ float awk[7];

    const float* Xbase = hs + (size_t)b * SS * 512;
    for (int d = tid; d < 512; d += 256) {
        Xs[d] = Xbase[(size_t)s * 512 + d];
#pragma unroll
        for (int k = 0; k < 7; ++k) {
            int idx = s + k - 3;
            idx = idx < 0 ? 0 : (idx > SS - 1 ? SS - 1 : idx);
            Xw[k][d] = Xbase[(size_t)idx * 512 + d];
        }
    }
    if (tid < 23) sums[tid] = 0.0f;
    __syncthreads();

    // partial sums: 0..6 dot(Xw_k, aw[0:512]); 7..13 dot(Xw_k, X);
    //               14..20 dot(Xw_k, Xw_k);   21 dot(X, aw[512:1024]); 22 dot(X,X)
    float part[23];
#pragma unroll
    for (int q = 0; q < 23; ++q) part[q] = 0.0f;
    for (int d = tid; d < 512; d += 256) {
        float xc = Xs[d];
        part[21] += xc * aw[512 + d];
        part[22] += xc * xc;
#pragma unroll
        for (int k = 0; k < 7; ++k) {
            float xw = Xw[k][d];
            part[k]      += xw * aw[d];
            part[7 + k]  += xw * xc;
            part[14 + k] += xw * xw;
        }
    }
#pragma unroll
    for (int q = 0; q < 23; ++q) {
        float v = part[q];
        for (int o = 16; o > 0; o >>= 1) v += __shfl_xor(v, o, 32);
        if ((tid & 31) == 0) atomicAdd(&sums[q], v);
    }
    __syncthreads();

    if (tid == 0) {
        const int dlb = dl[b];
        float nc = fmaxf(sqrtf(sums[22]), 1e-12f);
        float sc[7];
        float mx = -3.4e38f;
#pragma unroll
        for (int k = 0; k < 7; ++k) {
            int idx = s + k - 3;
            bool valid = (idx >= 0) && (idx < dlb) && (s < dlb);
            float nw = fmaxf(sqrtf(sums[14 + k]), 1e-12f);
            float sim = sigf(sums[7 + k] / (nw * nc));
            float sv = sums[k] + sums[21] + sim * aw[1024] + ab[0];
            sc[k] = valid ? sv : -1.0e30f;
            mx = fmaxf(mx, sc[k]);
        }
        float tot = 0.0f;
#pragma unroll
        for (int k = 0; k < 7; ++k) { sc[k] = __expf(sc[k] - mx); tot += sc[k]; }
        float inv = 1.0f / tot;
#pragma unroll
        for (int k = 0; k < 7; ++k) awk[k] = sc[k] * inv;
    }
    __syncthreads();

    const bool posvalid = s < dl[b];
    _Float16* o = hr + (size_t)token * 1024;
    for (int d = tid; d < 512; d += 256) {
        float ctx = 0.0f;
#pragma unroll
        for (int k = 0; k < 7; ++k) ctx += awk[k] * Xw[k][d];
        o[d]       = (_Float16)Xs[d];
        o[512 + d] = (_Float16)(posvalid ? ctx : 0.0f);
    }
}

// ---------------------------------------------------------------------------
// Classifier: logits = h2 @ cls_W^T + cls_b, softmax over 2 tags
// ---------------------------------------------------------------------------
__global__ __launch_bounds__(256) void cls_kernel(const float* __restrict__ h2,
                                                  const float* __restrict__ Wc,
                                                  const float* __restrict__ bc,
                                                  float* __restrict__ out)
{
    int token = blockIdx.x * blockDim.x + threadIdx.x;
    if (token >= BB * SS) return;
    const float* hv = h2 + (size_t)token * 512;
    float l0 = bc[0], l1 = bc[1];
    for (int d = 0; d < 512; ++d) {
        float h = hv[d];
        l0 += h * Wc[d];
        l1 += h * Wc[512 + d];
    }
    float mx = fmaxf(l0, l1);
    float e0 = __expf(l0 - mx), e1 = __expf(l1 - mx);
    float inv = 1.0f / (e0 + e1);
    out[token * 2 + 0] = e0 * inv;
    out[token * 2 + 1] = e1 * inv;
}

// ---------------------------------------------------------------------------
// Host orchestration
// ---------------------------------------------------------------------------
extern "C" void kernel_launch(void* const* d_in, const int* in_sizes, int n_in,
                              void* d_out, int out_size, void* d_ws, size_t ws_size,
                              hipStream_t stream)
{
    (void)in_sizes; (void)n_in; (void)out_size; (void)ws_size;

    const float* x     = (const float*)d_in[0];
    const int*   slen  = (const int*)d_in[1];
    const int*   dlen  = (const int*)d_in[2];
    const float* wl_Wf = (const float*)d_in[3];
    const float* wl_Uf = (const float*)d_in[4];
    const float* wl_bf = (const float*)d_in[5];
    const float* wl_Wb = (const float*)d_in[6];
    const float* wl_Ub = (const float*)d_in[7];
    const float* wl_bb = (const float*)d_in[8];
    const float* sl_Wf = (const float*)d_in[9];
    const float* sl_Uf = (const float*)d_in[10];
    const float* sl_bf = (const float*)d_in[11];
    const float* sl_Wb = (const float*)d_in[12];
    const float* sl_Ub = (const float*)d_in[13];
    const float* sl_bb = (const float*)d_in[14];
    const float* attnw = (const float*)d_in[15];
    const float* attnb = (const float*)d_in[16];
    const float* m2_Wf = (const float*)d_in[17];
    const float* m2_Uf = (const float*)d_in[18];
    const float* m2_bf = (const float*)d_in[19];
    const float* m2_Wb = (const float*)d_in[20];
    const float* m2_Ub = (const float*)d_in[21];
    const float* m2_bb = (const float*)d_in[22];
    const float* clsW  = (const float*)d_in[23];
    const float* clsB  = (const float*)d_in[24];
    float* out = (float*)d_out;

    char* ws = (char*)d_ws;
    size_t off = 0;
    auto take = [&](size_t bytes) -> char* {
        char* p = ws + off;
        off = (off + bytes + 255) & ~(size_t)255;
        return p;
    };

    _Float16* xh      = (_Float16*)take((size_t)BB * SS * WW * EPAD * 2);
    _Float16* wl_wB0  = (_Float16*)take((size_t)GG * EPAD * 2);
    _Float16* wl_wB1  = (_Float16*)take((size_t)GG * EPAD * 2);
    _Float16* wl_uB0  = (_Float16*)take((size_t)GG * 256 * 2);
    _Float16* wl_uB1  = (_Float16*)take((size_t)GG * 256 * 2);
    float*    wl_b0   = (float*)take((size_t)GG * 4);
    float*    wl_b1   = (float*)take((size_t)GG * 4);
    _Float16* sl_wB0  = (_Float16*)take((size_t)GG * 512 * 2);
    _Float16* sl_wB1  = (_Float16*)take((size_t)GG * 512 * 2);
    _Float16* sl_uB0  = (_Float16*)take((size_t)GG * 256 * 2);
    _Float16* sl_uB1  = (_Float16*)take((size_t)GG * 256 * 2);
    float*    sl_b0   = (float*)take((size_t)GG * 4);
    float*    sl_b1   = (float*)take((size_t)GG * 4);
    _Float16* m2_wB0  = (_Float16*)take((size_t)GG * 1024 * 2);
    _Float16* m2_wB1  = (_Float16*)take((size_t)GG * 1024 * 2);
    _Float16* m2_uB0  = (_Float16*)take((size_t)GG * 256 * 2);
    _Float16* m2_uB1  = (_Float16*)take((size_t)GG * 256 * 2);
    float*    m2_b0   = (float*)take((size_t)GG * 4);
    float*    m2_b1   = (float*)take((size_t)GG * 4);
    float*    pooled  = (float*)take((size_t)BB * SS * 512 * 4);
    _Float16* ph      = (_Float16*)take((size_t)BB * SS * 512 * 2);
    float*    hs      = (float*)take((size_t)BB * SS * 512 * 4);
    _Float16* hr      = (_Float16*)take((size_t)BB * SS * 1024 * 2);
    float*    h2      = (float*)take((size_t)BB * SS * 512 * 4);

    // ---- stage 0: conversions ----
    {
        size_t n = (size_t)BB * SS * WW * EPAD;
        cvt_x_kernel<<<(unsigned)((n + 255) / 256), 256, 0, stream>>>(x, xh);
    }
    cvt_w_kernel<<<GG, 256, 0, stream>>>(wl_Wf, wl_wB0, EE, EPAD);
    cvt_w_kernel<<<GG, 256, 0, stream>>>(wl_Wb, wl_wB1, EE, EPAD);
    cvt_w_kernel<<<GG, 256, 0, stream>>>(wl_Uf, wl_uB0, 256, 256);
    cvt_w_kernel<<<GG, 256, 0, stream>>>(wl_Ub, wl_uB1, 256, 256);
    cvt_b_kernel<<<4, 256, 0, stream>>>(wl_bf, wl_b0);
    cvt_b_kernel<<<4, 256, 0, stream>>>(wl_bb, wl_b1);
    cvt_w_kernel<<<GG, 256, 0, stream>>>(sl_Wf, sl_wB0, 512, 512);
    cvt_w_kernel<<<GG, 256, 0, stream>>>(sl_Wb, sl_wB1, 512, 512);
    cvt_w_kernel<<<GG, 256, 0, stream>>>(sl_Uf, sl_uB0, 256, 256);
    cvt_w_kernel<<<GG, 256, 0, stream>>>(sl_Ub, sl_uB1, 256, 256);
    cvt_b_kernel<<<4, 256, 0, stream>>>(sl_bf, sl_b0);
    cvt_b_kernel<<<4, 256, 0, stream>>>(sl_bb, sl_b1);
    cvt_w_kernel<<<GG, 256, 0, stream>>>(m2_Wf, m2_wB0, 1024, 1024);
    cvt_w_kernel<<<GG, 256, 0, stream>>>(m2_Wb, m2_wB1, 1024, 1024);
    cvt_w_kernel<<<GG, 256, 0, stream>>>(m2_Uf, m2_uB0, 256, 256);
    cvt_w_kernel<<<GG, 256, 0, stream>>>(m2_Ub, m2_uB1, 256, 256);
    cvt_b_kernel<<<4, 256, 0, stream>>>(m2_bf, m2_b0);
    cvt_b_kernel<<<4, 256, 0, stream>>>(m2_bb, m2_b1);

    // ---- stage 1: word-level BiLSTM + max-pool -> pooled (B*S, 512) ----
    lstm_kernel<EPAD><<<dim3(BB * SS / 16, 2), 256, 0, stream>>>(
        xh, WW, WW * EPAD,
        wl_wB0, wl_uB0, wl_b0, wl_wB1, wl_uB1, wl_b1,
        slen, 0, pooled, 0, 0, 512);

    // ---- stage 2: doc-length mask + f16 ----
    cvt_pool_kernel<<<(BB * SS * 512 + 255) / 256, 256, 0, stream>>>(pooled, dlen, ph);

    // ---- stage 3: sentence-level BiLSTM -> hs (B,S,512), masked ----
    lstm_kernel<512><<<dim3(1, 2), 256, 0, stream>>>(
        ph, SS, SS * 512,
        sl_wB0, sl_uB0, sl_b0, sl_wB1, sl_uB1, sl_b1,
        dlen, 1, hs, SS * 512, 512, 0);

    // ---- stage 4: restricted attention -> hr (B,S,1024) f16 ----
    attn_kernel<<<BB * SS, 256, 0, stream>>>(hs, dlen, attnw, attnb, hr);

    // ---- stage 5: m2 BiLSTM -> h2 (B,S,512) ----
    lstm_kernel<1024><<<dim3(1, 2), 256, 0, stream>>>(
        hr, SS, SS * 1024,
        m2_wB0, m2_uB0, m2_b0, m2_wB1, m2_uB1, m2_b1,
        nullptr, 1, h2, SS * 512, 512, 0);

    // ---- stage 6: classifier + softmax -> out (B,S,2) ----
    cls_kernel<<<(BB * SS + 255) / 256, 256, 0, stream>>>(h2, clsW, clsB, out);
}
